// WaveNet_32263794328279
// MI455X (gfx1250) — matrix operand order repaired and checked by reading, not compiled, not run on hardware
//
#include <hip/hip_runtime.h>

#define HID    128
#define NCLS   256
#define BATCH  4
#define TLEN   64000
#define PADT   512
#define TT     64          // time tile per block for dilated layers
#define CTT    256         // time tile per block for classifier
#define NLAYER 28
#define LROW   136         // LDS row stride in halves (128 data + 8 pad -> 4-bank skew)

typedef __attribute__((ext_vector_type(16))) _Float16 v16h;
typedef __attribute__((ext_vector_type(8)))  _Float16 v8h;
typedef __attribute__((ext_vector_type(8)))  float    v8f;
typedef __attribute__((ext_vector_type(4)))  int      v4i;

#define BSTRIDE ((size_t)(PADT + TLEN) * HID)   // halves per batch image

#ifndef __has_builtin
#define __has_builtin(x) 0
#endif
#if __has_builtin(__builtin_amdgcn_global_load_async_to_lds_b128)
#define HAVE_ASYNC_LDS 1
#else
#define HAVE_ASYNC_LDS 0
#endif

static __device__ __forceinline__ v16h load_v16h(const _Float16* p) {
  v8h lo = *(const v8h*)p;
  v8h hi = *(const v8h*)(p + 8);
  v16h r;
#pragma unroll
  for (int i = 0; i < 8; ++i) { r[i] = lo[i]; r[i + 8] = hi[i]; }
  return r;
}

// Async global -> LDS 16-byte copy (ASYNCcnt path). Builtin signature (leaked
// by clang diagnostics): (v4i __device__* src, v4i __shared__* dst, i32 off,
// i32 cpol). Cast through integers to produce the address-space pointers.
static __device__ __forceinline__ void lds_copy16(_Float16* larr, unsigned lidx,
                                                  const _Float16* gp) {
#if HAVE_ASYNC_LDS
  __builtin_amdgcn_global_load_async_to_lds_b128(
      (__attribute__((address_space(1))) v4i*)(unsigned long long)(size_t)gp,
      (__attribute__((address_space(3))) v4i*)(unsigned)(size_t)(larr + lidx),
      0, 0);
#else
  // VDST = LDS byte address, VADDR = 64-bit global address (ISA 08 §4.3).
  asm volatile("global_load_async_to_lds_b128 %0, %1, off"
               :: "v"((unsigned)(size_t)(larr + lidx)), "v"(gp) : "memory");
#endif
}

static __device__ __forceinline__ void async_wait0() {
#if __has_builtin(__builtin_amdgcn_s_wait_asynccnt)
  __builtin_amdgcn_s_wait_asynccnt(0);
#else
  asm volatile("s_wait_asynccnt 0x0" ::: "memory");
#endif
}

// ---------------------------------------------------------------------------
// Pack rest-layer weights [27][co=128][ci=128][tap=2] f32 into WMMA A-fragment
// order: [layer][m=8][ks=8][lane=32][half=16] f16 (k = ks*32 + 16*(lane/16) + half, contiguous per lane half).
// ---------------------------------------------------------------------------
__global__ void pack_w_rest_kernel(const float* __restrict__ w_rest,
                                   _Float16* __restrict__ wf) {
  int idx = blockIdx.x * blockDim.x + threadIdx.x;   // 27 * 32768
  int h     = idx & 15;
  int lane  = (idx >> 4) & 31;
  int ks    = (idx >> 9) & 7;
  int m     = (idx >> 12) & 7;
  int layer = idx >> 15;
  if (layer >= NLAYER - 1) return;
  int klocal = h + ((lane >> 4) << 4);
  int k   = ks * 32 + klocal;          // 0..255
  int tap = k >> 7;
  int ci  = k & 127;
  int co  = m * 16 + (lane & 15);
  wf[idx] = (_Float16)w_rest[(((size_t)layer * HID + co) * HID + ci) * 2 + tap];
}

// Classifier weights [o=256][ci=128] -> [m=16][ks=4][lane][half] f16.
__global__ void pack_w_cls_kernel(const float* __restrict__ w_cls,
                                  _Float16* __restrict__ wf) {
  int idx = blockIdx.x * blockDim.x + threadIdx.x;   // 32768
  int h    = idx & 15;
  int lane = (idx >> 4) & 31;
  int ks   = (idx >> 9) & 3;
  int m    = idx >> 11;
  if (m >= 16) return;
  int klocal = h + ((lane >> 4) << 4);
  int k  = ks * 32 + klocal;           // ci
  int co = m * 16 + (lane & 15);
  wf[idx] = (_Float16)w_cls[(size_t)co * HID + k];
}

// Zero the causal pad region of an activation buffer (BATCH * PADT * HID halves).
__global__ void zero_pad_kernel(_Float16* __restrict__ buf) {
  int idx = blockIdx.x * blockDim.x + threadIdx.x;   // 32768 (x8 halves each)
  int n = idx >> 13;                                 // PADT*HID/8 = 8192
  int w = idx & 8191;
  v8h z = {};
  *(v8h*)(buf + (size_t)n * BSTRIDE + (size_t)w * 8) = z;
}

// Layer 0: 1 input channel, d=1. y[n][t][co] = relu(w0[co]*x[t-1] + w1[co]*x[t])
__global__ void layer0_kernel(const float* __restrict__ seq,
                              const float* __restrict__ w_first,
                              _Float16* __restrict__ out) {
  int idx = blockIdx.x * blockDim.x + threadIdx.x;   // TLEN*16 per batch
  int n  = blockIdx.y;
  int cg = idx & 15;                                 // channel group of 8
  int t  = idx >> 4;
  float x1 = seq[(size_t)n * TLEN + t];
  float x0 = (t > 0) ? seq[(size_t)n * TLEN + t - 1] : 0.0f;
  v8h o;
#pragma unroll
  for (int r = 0; r < 8; ++r) {
    int co = cg * 8 + r;
    float y = w_first[co * 2 + 0] * x0 + w_first[co * 2 + 1] * x1;
    o[r] = (_Float16)(y > 0.0f ? y : 0.0f);
  }
  *(v8h*)(out + (size_t)n * BSTRIDE + ((size_t)(PADT + t) * HID + cg * 8)) = o;
}

// ---------------------------------------------------------------------------
// Dilated conv layer as WMMA GEMM: M=128 (8 waves x 16), K=256 (8 x 32),
// N = TT time steps. Both tap regions are staged once per block into LDS via
// async global->LDS b128 copies; WMMA B operands then come from ds_load_b128
// (row stride 272B -> 4-bank skew per lane -> conflict-free 16-lane gathers).
// ---------------------------------------------------------------------------
__global__ __launch_bounds__(256) void wmma_layer_kernel(
    const _Float16* __restrict__ xin, _Float16* __restrict__ xout,
    const _Float16* __restrict__ wf, int d) {
  __shared__ _Float16 lds[2 * TT * LROW];   // 34,816 bytes
  const int tid   = threadIdx.x;
  const int lane  = tid & 31;
  const int wave  = tid >> 5;               // m-tile 0..7
  const int n     = blockIdx.y;
  const int tbase = blockIdx.x * TT;
  const _Float16* xn = xin  + (size_t)n * BSTRIDE;
  _Float16*       yn = xout + (size_t)n * BSTRIDE;

  // A fragments: resident in registers for the whole block.
  v16h A[8];
#pragma unroll
  for (int ks = 0; ks < 8; ++ks)
    A[ks] = load_v16h(wf + (((wave * 8 + ks) * 32 + lane) << 4));

  // Stage region 0 (x[t-d]) and region 1 (x[t]) into LDS: 2048 b128 chunks.
#pragma unroll
  for (int i = 0; i < 8; ++i) {
    int idx    = tid + i * 256;             // 0..2047
    int region = idx >> 10;                 // 1024 chunks per region
    int row    = (idx >> 4) & (TT - 1);
    int chunk  = idx & 15;
    int tsh    = region ? 0 : d;
    const _Float16* gp = xn + (size_t)(PADT + tbase + row - tsh) * HID + chunk * 8;
    lds_copy16(lds, (unsigned)((region * TT + row) * LROW + chunk * 8), gp);
  }
  async_wait0();
  __syncthreads();

  const int tl = lane & 15;                 // N within tile
  const int hi = lane >> 4;                 // lane half select
  const int cobase = wave * 16 + (hi << 3); // 8 consecutive out channels

  for (int nt = 0; nt < TT / 16; nt += 2) {
    const int t0l = nt * 16;
    v8f acc0 = {}; v8f acc1 = {};
#pragma unroll
    for (int ks = 0; ks < 8; ++ks) {
      const int region = (ks < 4) ? 0 : 1;  // tap0 uses x[t-d]
      const int cib = ((ks & 3) << 5) + (hi << 4);
      const _Float16* bp = &lds[(region * TT + t0l + tl) * LROW + cib];
      v16h B0 = load_v16h(bp);
      v16h B1 = load_v16h(bp + 16 * LROW);
      acc0 = __builtin_amdgcn_wmma_f32_16x16x32_f16(false, A[ks], false, B0,
                                                    (short)0, acc0, false, false);
      acc1 = __builtin_amdgcn_wmma_f32_16x16x32_f16(false, A[ks], false, B1,
                                                    (short)0, acc1, false, false);
    }
    v8h o0, o1;
#pragma unroll
    for (int r = 0; r < 8; ++r) {
      float y0 = acc0[r] > 0.0f ? acc0[r] : 0.0f;
      float y1 = acc1[r] > 0.0f ? acc1[r] : 0.0f;
      o0[r] = (_Float16)y0;
      o1[r] = (_Float16)y1;
    }
    _Float16* op = yn + (size_t)(PADT + tbase + t0l + tl) * HID + cobase;
    *(v8h*)op = o0;
    *(v8h*)(op + 16 * HID) = o1;
  }
}

// ---------------------------------------------------------------------------
// Classifier: out[n][o][t] = wcls[o,:] . x[n][t][:] + b[o]; M=256 (8 waves x
// 2 m-tiles), K=128 (4 x 32). Bias folded into accumulator init.
// ---------------------------------------------------------------------------
__global__ __launch_bounds__(256) void wmma_cls_kernel(
    const _Float16* __restrict__ xin, const _Float16* __restrict__ wf,
    const float* __restrict__ b_cls, float* __restrict__ out) {
  const int lane  = threadIdx.x & 31;
  const int wave  = threadIdx.x >> 5;
  const int n     = blockIdx.y;
  const int tbase = blockIdx.x * CTT;
  const _Float16* xn = xin + (size_t)n * BSTRIDE;

  const int tl = lane & 15;
  const int hi = lane >> 4;
  const int ob0 = wave * 16 + (hi << 3);
  const int ob1 = (wave + 8) * 16 + (hi << 3);

  v16h A0[4], A1[4];
#pragma unroll
  for (int ks = 0; ks < 4; ++ks) {
    A0[ks] = load_v16h(wf + (((wave * 4 + ks) * 32 + lane) << 4));
    A1[ks] = load_v16h(wf + ((((wave + 8) * 4 + ks) * 32 + lane) << 4));
  }
  float bias0[8], bias1[8];
#pragma unroll
  for (int r = 0; r < 8; ++r) { bias0[r] = b_cls[ob0 + r]; bias1[r] = b_cls[ob1 + r]; }

  for (int nt = 0; nt < CTT / 16; ++nt) {
    const int t0 = tbase + nt * 16;
    v8f acc0, acc1;
#pragma unroll
    for (int r = 0; r < 8; ++r) { acc0[r] = bias0[r]; acc1[r] = bias1[r]; }
#pragma unroll
    for (int ks = 0; ks < 4; ++ks) {
      const int cib = (ks << 5) + (hi << 4);
      v16h Bf = load_v16h(xn + (size_t)(PADT + t0 + tl) * HID + cib);
      acc0 = __builtin_amdgcn_wmma_f32_16x16x32_f16(false, A0[ks], false, Bf,
                                                    (short)0, acc0, false, false);
      acc1 = __builtin_amdgcn_wmma_f32_16x16x32_f16(false, A1[ks], false, Bf,
                                                    (short)0, acc1, false, false);
    }
    const int t = t0 + tl;
#pragma unroll
    for (int r = 0; r < 8; ++r) {
      out[((size_t)n * NCLS + ob0 + r) * TLEN + t] = acc0[r];
      out[((size_t)n * NCLS + ob1 + r) * TLEN + t] = acc1[r];
    }
  }
}

extern "C" void kernel_launch(void* const* d_in, const int* in_sizes, int n_in,
                              void* d_out, int out_size, void* d_ws, size_t ws_size,
                              hipStream_t stream) {
  const float* seq     = (const float*)d_in[0];
  const float* w_first = (const float*)d_in[1];
  const float* w_rest  = (const float*)d_in[2];
  const float* w_cls   = (const float*)d_in[3];
  const float* b_cls   = (const float*)d_in[4];
  float* out = (float*)d_out;
  (void)in_sizes; (void)n_in; (void)out_size; (void)ws_size;

  char* ws = (char*)d_ws;
  const size_t bufBytes = BSTRIDE * (size_t)BATCH * 2;            // f16 halves * 2B
  _Float16* buf0 = (_Float16*)ws;
  _Float16* buf1 = (_Float16*)(ws + bufBytes);
  _Float16* wfr  = (_Float16*)(ws + 2 * bufBytes);                // 27*32768 halves
  _Float16* wfc  = (_Float16*)(ws + 2 * bufBytes + (size_t)(NLAYER - 1) * 32768 * 2);

  pack_w_rest_kernel<<<(NLAYER - 1) * 128, 256, 0, stream>>>(w_rest, wfr);
  pack_w_cls_kernel<<<128, 256, 0, stream>>>(w_cls, wfc);
  zero_pad_kernel<<<128, 256, 0, stream>>>(buf0);
  zero_pad_kernel<<<128, 256, 0, stream>>>(buf1);
  layer0_kernel<<<dim3(TLEN * 16 / 256, BATCH), 256, 0, stream>>>(seq, w_first, buf0);

  _Float16* cur = buf0;
  _Float16* nxt = buf1;
  for (int i = 1; i < NLAYER; ++i) {
    int l = i % 14;
    int d = 1 << (l % 10);
    wmma_layer_kernel<<<dim3(TLEN / TT, BATCH), 256, 0, stream>>>(
        cur, nxt, wfr + (size_t)(i - 1) * 32768, d);
    _Float16* t = cur; cur = nxt; nxt = t;
  }
  wmma_cls_kernel<<<dim3(TLEN / CTT, BATCH), 256, 0, stream>>>(cur, wfc, b_cls, out);
}
